// Model_25847113188125
// MI455X (gfx1250) — matrix operand order repaired and checked by reading, not compiled, not run on hardware
//
#include <hip/hip_runtime.h>
#include <hip/hip_bf16.h>

#define NB   4
#define CHN  1024
#define NH   16
#define HD   64
#define SEQ  1024
#define BSZ  2
#define ROWS (BSZ*SEQ)   // 2048

typedef __attribute__((ext_vector_type(16))) __bf16        v16bf;
typedef __attribute__((ext_vector_type(8)))  float         v8f;
typedef __attribute__((ext_vector_type(4)))  unsigned int  u32x4;
typedef __attribute__((ext_vector_type(2)))  unsigned int  u32x2;
typedef unsigned short ushort;

union Frag16 { v16bf v; u32x4 q[2]; };

__device__ __forceinline__ unsigned short f2bf_sw(float f) {
  union { float f; unsigned u; } x; x.f = f;
  unsigned r = x.u + 0x7fffu + ((x.u >> 16) & 1u);
  return (unsigned short)(r >> 16);
}
#if defined(__has_builtin) && __has_builtin(__builtin_amdgcn_cvt_pk_bf16_f32)
__device__ __forceinline__ unsigned pack2bf(float a, float b) {
  typedef __bf16 v2bf __attribute__((ext_vector_type(2)));
  union { v2bf v; unsigned u; } x;
  x.v = __builtin_amdgcn_cvt_pk_bf16_f32(a, b);
  return x.u;
}
#else
__device__ __forceinline__ unsigned pack2bf(float a, float b) {
  return (unsigned)f2bf_sw(a) | ((unsigned)f2bf_sw(b) << 16);
}
#endif

// CDNA5 async global->LDS copy (ASYNCcnt) ------------------------------------
__device__ __forceinline__ void async_copy16(void* lds, const void* g) {
  unsigned ldsa = (unsigned)(unsigned long long)lds;          // low 32 bits = LDS offset
  unsigned long long ga = (unsigned long long)g;
  asm volatile("global_load_async_to_lds_b128 %0, %1, off" :: "v"(ldsa), "v"(ga) : "memory");
}
__device__ __forceinline__ void wait_async() {
  asm volatile("s_wait_asynccnt 0" ::: "memory");
}

// ---------------------------------------------------------------------------
// Weight prep: W[K][N] f32 -> Wt[N][K] bf16 (LDS-tiled transpose + cvt_pk).
// ---------------------------------------------------------------------------
__global__ __launch_bounds__(256)
void transpose_f32_to_bf16(const float* __restrict__ W, ushort* __restrict__ Wt,
                           int K, int N)
{
  __shared__ float tile[32][33];
  int k0 = blockIdx.x * 32, n0 = blockIdx.y * 32;
  int tx = threadIdx.x & 31, ty = threadIdx.x >> 5;    // ty 0..7
#pragma unroll
  for (int i = 0; i < 32; i += 8)
    tile[ty + i][tx] = W[(long)(k0 + ty + i) * N + n0 + tx];
  __syncthreads();
  if (tx < 16) {
#pragma unroll
    for (int i = 0; i < 32; i += 8) {
      int n = n0 + ty + i;
      unsigned pk = pack2bf(tile[2 * tx][ty + i], tile[2 * tx + 1][ty + i]);
      *reinterpret_cast<unsigned*>(&Wt[(long)n * K + k0 + 2 * tx]) = pk;
    }
  }
}

// ---------------------------------------------------------------------------
// GEMM: out = act(A[M,K]bf16 @ Bt[N,K]bf16^T + bias) (+Res).  f32 accum.
// Tile 64x128x32, double-buffered LDS, async-to-LDS staging, 8 waves=4mx2n,
// each wave 16x64 (4 WMMAs/k-step).  M%64==0, N%128==0, K%32==0.
// ---------------------------------------------------------------------------
__global__ __launch_bounds__(256)
void gemm_bf16(const ushort* __restrict__ A, const ushort* __restrict__ Bt,
               const float* __restrict__ bias, const float* __restrict__ Res,
               float* __restrict__ Cout, ushort* __restrict__ OutBf,
               int M, int N, int K, int fuse_gelu)
{
  __shared__ ushort As[2][64 * 32];    // [m][k]
  __shared__ ushort Bs[2][128 * 32];   // [n][k]
  const int t = threadIdx.x;
  const int wid = t >> 5, lane = t & 31;
  const int m0 = blockIdx.x * 64;
  const int n0 = blockIdx.y * 128;
  const int wm = wid & 3, wn = wid >> 2;
  v8f acc[4] = {v8f{}, v8f{}, v8f{}, v8f{}};
  const int arow  = wm * 16 + (lane & 15);
  const int akoff = (lane < 16) ? 0 : 16;
  const int bkoff = (lane < 16) ? 0 : 16;
  const int bn    = wn * 64 + (lane & 15);
  // per-thread staging chunk coords (16B chunks)
  const int ar = t >> 2,  ac = (t & 3) * 8;       // A: 256 chunks

  auto issue = [&](int k0, int buf) {
    async_copy16(&As[buf][ar * 32 + ac], &A[(long)(m0 + ar) * K + k0 + ac]);
#pragma unroll
    for (int e = 0; e < 2; ++e) {
      int p = t + e * 256;                        // B: 512 chunks
      int n = p >> 2, c = (p & 3) * 8;
      async_copy16(&Bs[buf][n * 32 + c], &Bt[(long)(n0 + n) * K + k0 + c]);
    }
  };

  issue(0, 0);
  wait_async();
  __syncthreads();
  const int nk = K >> 5;
  for (int ki = 0; ki < nk; ++ki) {
    const int buf = ki & 1;
    if (ki + 1 < nk) issue((ki + 1) << 5, buf ^ 1);
    Frag16 a;
    a.q[0] = *reinterpret_cast<const u32x4*>(&As[buf][arow * 32 + akoff]);
    a.q[1] = *reinterpret_cast<const u32x4*>(&As[buf][arow * 32 + 8 + akoff]);
#pragma unroll
    for (int nt = 0; nt < 4; ++nt) {
      Frag16 b;
      b.q[0] = *reinterpret_cast<const u32x4*>(&Bs[buf][(bn + nt * 16) * 32 + bkoff]);
      b.q[1] = *reinterpret_cast<const u32x4*>(&Bs[buf][(bn + nt * 16) * 32 + bkoff + 8]);
      acc[nt] = __builtin_amdgcn_wmma_f32_16x16x32_bf16(false, a.v, false, b.v,
                                                        (short)0, acc[nt], false, false);
    }
    wait_async();
    __syncthreads();
  }

#pragma unroll
  for (int nt = 0; nt < 4; ++nt) {
    int gn = n0 + wn * 64 + nt * 16 + (lane & 15);
    float bs = bias ? bias[gn] : 0.f;
#pragma unroll
    for (int r = 0; r < 8; ++r) {
      int gm = m0 + wm * 16 + r + ((lane < 16) ? 0 : 8);
      float v = acc[nt][r] + bs;
      if (fuse_gelu) v = 0.5f * v * (1.f + erff(v * 0.70710678f));
      if (Res) v += Res[(long)gm * N + gn];
      if (OutBf) {                       // bf16 out: pair-pack via lane^1
        float pv = __shfl_xor(v, 1, 32);
        if ((lane & 1) == 0)
          *reinterpret_cast<unsigned*>(&OutBf[(long)gm * N + gn]) = pack2bf(v, pv);
      } else {
        Cout[(long)gm * N + gn] = v;
      }
    }
  }
}

// ---------------------------------------------------------------------------
// Sincos 2D positional embedding, computed on the fly (f32 out).
// ---------------------------------------------------------------------------
__global__ void add_pos_embed(const float* __restrict__ X, float* __restrict__ Y)
{
  int row = blockIdx.x;          // b*SEQ + l
  int l  = row & (SEQ - 1);
  int yy = l >> 5, xx = l & 31;
  for (int c = threadIdx.x; c < CHN; c += blockDim.x) {
    int half = c >> 9, cc = c & 511;
    float pos   = half ? (float)yy : (float)xx;
    float omega = __powf(10000.f, -(float)(cc & 255) * (1.f / 256.f));
    float a     = pos * omega;
    float pe    = (cc < 256) ? __sinf(a) : __cosf(a);
    Y[(long)row * CHN + c] = X[(long)row * CHN + c] + pe;
  }
}

// ---------------------------------------------------------------------------
// LayerNorm over last dim (1024): wave per row, float4 in, bf16-packed out.
// ---------------------------------------------------------------------------
__global__ __launch_bounds__(256)
void layernorm_rows(const float* __restrict__ X, const float* __restrict__ g,
                    const float* __restrict__ bta, ushort* __restrict__ Yb,
                    int nrows, float eps)
{
  int wid = threadIdx.x >> 5, lane = threadIdx.x & 31;
  int row = blockIdx.x * 8 + wid;
  if (row >= nrows) return;
  const float4* xr = reinterpret_cast<const float4*>(X + (long)row * CHN);
  float4 v[8];
  float s = 0.f, s2 = 0.f;
#pragma unroll
  for (int i = 0; i < 8; ++i) {
    float4 q = xr[lane + i * 32];
    v[i] = q;
    s  += q.x + q.y + q.z + q.w;
    s2 += q.x*q.x + q.y*q.y + q.z*q.z + q.w*q.w;
  }
#pragma unroll
  for (int m = 1; m < 32; m <<= 1) { s += __shfl_xor(s, m, 32); s2 += __shfl_xor(s2, m, 32); }
  float mean = s * (1.f / CHN);
  float var  = s2 * (1.f / CHN) - mean * mean;
  float inv  = rsqrtf(var + eps);
#pragma unroll
  for (int i = 0; i < 8; ++i) {
    int c4 = lane + i * 32;
    float4 q = v[i];
    q.x = (q.x - mean) * inv; q.y = (q.y - mean) * inv;
    q.z = (q.z - mean) * inv; q.w = (q.w - mean) * inv;
    if (g) {
      float4 gg = reinterpret_cast<const float4*>(g)[c4];
      float4 bb = reinterpret_cast<const float4*>(bta)[c4];
      q.x = q.x * gg.x + bb.x; q.y = q.y * gg.y + bb.y;
      q.z = q.z * gg.z + bb.z; q.w = q.w * gg.w + bb.w;
    }
    u32x2 pk; pk[0] = pack2bf(q.x, q.y); pk[1] = pack2bf(q.z, q.w);
    *reinterpret_cast<u32x2*>(&Yb[(long)row * CHN + 4 * c4]) = pk;
  }
}

// ---------------------------------------------------------------------------
// Split qkv f32 (2048x3072) -> head-major bf16 q/k/v [b,h,l,d], QK-norm d=64.
// Wave per (row, head); lane covers dd=lane, lane+32; bf16 pair-packed stores.
// ---------------------------------------------------------------------------
__global__ __launch_bounds__(256)
void qkv_split_qknorm(const float* __restrict__ QKV,
                      const float* __restrict__ nqg, const float* __restrict__ nqb,
                      const float* __restrict__ nkg, const float* __restrict__ nkb,
                      ushort* __restrict__ Qb, ushort* __restrict__ Kb,
                      ushort* __restrict__ Vb)
{
  int wid = threadIdx.x >> 5, lane = threadIdx.x & 31;
  int task = blockIdx.x * 8 + wid;          // row * NH + h
  int row = task >> 4, h = task & 15;
  if (row >= ROWS) return;
  const float* base = QKV + (long)row * 3 * CHN + h * HD;
  float q0 = base[lane],           q1 = base[lane + 32];
  float k0 = base[CHN + lane],     k1 = base[CHN + lane + 32];
  float v0 = base[2*CHN + lane],   v1 = base[2*CHN + lane + 32];
  float qs = q0 + q1, qs2 = q0*q0 + q1*q1;
  float ks = k0 + k1, ks2 = k0*k0 + k1*k1;
#pragma unroll
  for (int m = 1; m < 32; m <<= 1) {
    qs  += __shfl_xor(qs,  m, 32); qs2 += __shfl_xor(qs2, m, 32);
    ks  += __shfl_xor(ks,  m, 32); ks2 += __shfl_xor(ks2, m, 32);
  }
  float qm = qs * (1.f/HD), qv = qs2 * (1.f/HD) - qm*qm, qi = rsqrtf(qv + 1e-5f);
  float km = ks * (1.f/HD), kv = ks2 * (1.f/HD) - km*km, ki = rsqrtf(kv + 1e-5f);
  float qn0 = (q0 - qm) * qi * nqg[lane]      + nqb[lane];
  float qn1 = (q1 - qm) * qi * nqg[lane + 32] + nqb[lane + 32];
  float kn0 = (k0 - km) * ki * nkg[lane]      + nkb[lane];
  float kn1 = (k1 - km) * ki * nkg[lane + 32] + nkb[lane + 32];
  float pq0 = __shfl_xor(qn0, 1, 32), pq1 = __shfl_xor(qn1, 1, 32);
  float pk0 = __shfl_xor(kn0, 1, 32), pk1 = __shfl_xor(kn1, 1, 32);
  float pv0 = __shfl_xor(v0,  1, 32), pv1 = __shfl_xor(v1,  1, 32);
  int b = row >> 10, l = row & 1023;
  long dst = (((long)b * NH + h) * SEQ + l) * HD;
  if ((lane & 1) == 0) {
    *reinterpret_cast<unsigned*>(&Qb[dst + lane])      = pack2bf(qn0, pq0);
    *reinterpret_cast<unsigned*>(&Qb[dst + lane + 32]) = pack2bf(qn1, pq1);
    *reinterpret_cast<unsigned*>(&Kb[dst + lane])      = pack2bf(kn0, pk0);
    *reinterpret_cast<unsigned*>(&Kb[dst + lane + 32]) = pack2bf(kn1, pk1);
    *reinterpret_cast<unsigned*>(&Vb[dst + lane])      = pack2bf(v0,  pv0);
    *reinterpret_cast<unsigned*>(&Vb[dst + lane + 32]) = pack2bf(v1,  pv1);
  }
}

// ---------------------------------------------------------------------------
// Fused flash attention, all-bf16 operands.  Block = (b,h) x 128 q-rows;
// wave = 16 q-rows.  Q/K async-copied to LDS; V transposed [d][key] with u16
// gathers.  Online softmax via wave-half shuffles; bf16 pair-packed output.
// ---------------------------------------------------------------------------
__global__ __launch_bounds__(256)
void flash_attn(const ushort* __restrict__ Qb, const ushort* __restrict__ Kb,
                const ushort* __restrict__ Vb, ushort* __restrict__ Attb)
{
  __shared__ ushort Qs[128 * 64];   // [qrow][d]
  __shared__ ushort Ks[32 * 64];    // [key][d]
  __shared__ ushort Vt[64 * 32];    // [d][key]
  __shared__ ushort Ps[8][16 * 32]; // per-wave P scratch [row][key]
  const int t = threadIdx.x, wid = t >> 5, lane = t & 31;
  const int bh = blockIdx.y;
  const int q0blk = blockIdx.x * 128;
  const long baseQ = (long)bh * SEQ * HD;

  // async-stage Q tile: 1024 x 16B chunks, 4/thread
#pragma unroll
  for (int e = 0; e < 4; ++e) {
    int p = t + e * 256;
    int r = p >> 3, c = (p & 7) * 8;
    async_copy16(&Qs[r * 64 + c], &Qb[baseQ + (long)(q0blk + r) * HD + c]);
  }
  wait_async();
  __syncthreads();

  const int wq    = wid * 16;
  const int arow  = wq + (lane & 15);
  const int akoff = (lane < 16) ? 0 : 16;
  const int bkoff = (lane < 16) ? 0 : 16;
  Frag16 qa0, qa1;
  qa0.q[0] = *reinterpret_cast<const u32x4*>(&Qs[arow * 64 + akoff]);
  qa0.q[1] = *reinterpret_cast<const u32x4*>(&Qs[arow * 64 + 8 + akoff]);
  qa1.q[0] = *reinterpret_cast<const u32x4*>(&Qs[arow * 64 + 32 + akoff]);
  qa1.q[1] = *reinterpret_cast<const u32x4*>(&Qs[arow * 64 + 40 + akoff]);

  float mrow[8], lrow[8];
  v8f o0 = {}, o1 = {}, o2 = {}, o3 = {};
#pragma unroll
  for (int r = 0; r < 8; ++r) { mrow[r] = -1e30f; lrow[r] = 0.f; }
  const float scale = 0.125f;

  for (int kb = 0; kb < SEQ; kb += 32) {
    __syncthreads();
    { // async K tile: 256 chunks, 1/thread
      int r = t >> 3, c = (t & 7) * 8;
      async_copy16(&Ks[r * 64 + c], &Kb[baseQ + (long)(kb + r) * HD + c]);
    }
#pragma unroll
    for (int e = 0; e < 4; ++e) {       // V transpose: u16 gathers -> b32 LDS
      int p = t + e * 256;
      int d = p & 63, ky = (p >> 6) * 2;
      unsigned lo = Vb[baseQ + (long)(kb + ky)     * HD + d];
      unsigned hi = Vb[baseQ + (long)(kb + ky + 1) * HD + d];
      *reinterpret_cast<unsigned*>(&Vt[d * 32 + ky]) = lo | (hi << 16);
    }
    wait_async();
    __syncthreads();

    // S = Q @ K^T
    int keyA = (lane & 15) * 64, keyB = keyA + 16 * 64;
    Frag16 bk00, bk01, bk10, bk11;
    bk00.q[0] = *reinterpret_cast<const u32x4*>(&Ks[keyA + bkoff]);
    bk00.q[1] = *reinterpret_cast<const u32x4*>(&Ks[keyA + bkoff + 8]);
    bk01.q[0] = *reinterpret_cast<const u32x4*>(&Ks[keyA + 32 + bkoff]);
    bk01.q[1] = *reinterpret_cast<const u32x4*>(&Ks[keyA + 32 + bkoff + 8]);
    bk10.q[0] = *reinterpret_cast<const u32x4*>(&Ks[keyB + bkoff]);
    bk10.q[1] = *reinterpret_cast<const u32x4*>(&Ks[keyB + bkoff + 8]);
    bk11.q[0] = *reinterpret_cast<const u32x4*>(&Ks[keyB + 32 + bkoff]);
    bk11.q[1] = *reinterpret_cast<const u32x4*>(&Ks[keyB + 32 + bkoff + 8]);
    v8f s0 = {}, s1 = {};
    s0 = __builtin_amdgcn_wmma_f32_16x16x32_bf16(false, qa0.v, false, bk00.v, (short)0, s0, false, false);
    s0 = __builtin_amdgcn_wmma_f32_16x16x32_bf16(false, qa1.v, false, bk01.v, (short)0, s0, false, false);
    s1 = __builtin_amdgcn_wmma_f32_16x16x32_bf16(false, qa0.v, false, bk10.v, (short)0, s1, false, false);
    s1 = __builtin_amdgcn_wmma_f32_16x16x32_bf16(false, qa1.v, false, bk11.v, (short)0, s1, false, false);

    float pn0[8], pn1[8];
#pragma unroll
    for (int r = 0; r < 8; ++r) {
      float x0 = s0[r] * scale, x1 = s1[r] * scale;
      float mx = fmaxf(x0, x1);
      mx = fmaxf(mx, __shfl_xor(mx, 1, 32));
      mx = fmaxf(mx, __shfl_xor(mx, 2, 32));
      mx = fmaxf(mx, __shfl_xor(mx, 4, 32));
      mx = fmaxf(mx, __shfl_xor(mx, 8, 32));
      float mnew = fmaxf(mrow[r], mx);
      float corr = __expf(mrow[r] - mnew);
      float p0 = __expf(x0 - mnew), p1 = __expf(x1 - mnew);
      float ps = p0 + p1;
      ps += __shfl_xor(ps, 1, 32); ps += __shfl_xor(ps, 2, 32);
      ps += __shfl_xor(ps, 4, 32); ps += __shfl_xor(ps, 8, 32);
      lrow[r] = lrow[r] * corr + ps;
      mrow[r] = mnew;
      o0[r] *= corr; o1[r] *= corr; o2[r] *= corr; o3[r] *= corr;
      pn0[r] = p0; pn1[r] = p1;
    }

    // D-layout -> A-layout via per-wave LDS (pair-packed b32 stores)
#pragma unroll
    for (int r = 0; r < 8; ++r) {
      int prow = r + ((lane < 16) ? 0 : 8);
      float a0 = pn0[r], a1 = pn1[r];
      float b0p = __shfl_xor(a0, 1, 32), b1p = __shfl_xor(a1, 1, 32);
      if ((lane & 1) == 0) {
        *reinterpret_cast<unsigned*>(&Ps[wid][prow * 32 + (lane & 15)])      = pack2bf(a0, b0p);
        *reinterpret_cast<unsigned*>(&Ps[wid][prow * 32 + 16 + (lane & 15)]) = pack2bf(a1, b1p);
      }
    }
    asm volatile("s_wait_dscnt 0" ::: "memory");

    Frag16 pa;
    pa.q[0] = *reinterpret_cast<const u32x4*>(&Ps[wid][(lane & 15) * 32 + akoff]);
    pa.q[1] = *reinterpret_cast<const u32x4*>(&Ps[wid][(lane & 15) * 32 + 8 + akoff]);
    Frag16 vf0, vf1, vf2, vf3;
    int d0 = (lane & 15) * 32;
    vf0.q[0] = *reinterpret_cast<const u32x4*>(&Vt[d0 + bkoff]);
    vf0.q[1] = *reinterpret_cast<const u32x4*>(&Vt[d0 + bkoff + 8]);
    vf1.q[0] = *reinterpret_cast<const u32x4*>(&Vt[d0 + 16*32 + bkoff]);
    vf1.q[1] = *reinterpret_cast<const u32x4*>(&Vt[d0 + 16*32 + bkoff + 8]);
    vf2.q[0] = *reinterpret_cast<const u32x4*>(&Vt[d0 + 32*32 + bkoff]);
    vf2.q[1] = *reinterpret_cast<const u32x4*>(&Vt[d0 + 32*32 + bkoff + 8]);
    vf3.q[0] = *reinterpret_cast<const u32x4*>(&Vt[d0 + 48*32 + bkoff]);
    vf3.q[1] = *reinterpret_cast<const u32x4*>(&Vt[d0 + 48*32 + bkoff + 8]);
    o0 = __builtin_amdgcn_wmma_f32_16x16x32_bf16(false, pa.v, false, vf0.v, (short)0, o0, false, false);
    o1 = __builtin_amdgcn_wmma_f32_16x16x32_bf16(false, pa.v, false, vf1.v, (short)0, o1, false, false);
    o2 = __builtin_amdgcn_wmma_f32_16x16x32_bf16(false, pa.v, false, vf2.v, (short)0, o2, false, false);
    o3 = __builtin_amdgcn_wmma_f32_16x16x32_bf16(false, pa.v, false, vf3.v, (short)0, o3, false, false);
  }

  const int b = bh >> 4, h = bh & 15;
#pragma unroll
  for (int r = 0; r < 8; ++r) {
    int q = q0blk + wq + r + ((lane < 16) ? 0 : 8);
    float inv = 1.f / lrow[r];
    long dst = ((long)(b * SEQ + q)) * CHN + h * HD + (lane & 15);
    float w0 = o0[r]*inv, w1 = o1[r]*inv, w2 = o2[r]*inv, w3 = o3[r]*inv;
    float p0 = __shfl_xor(w0,1,32), p1 = __shfl_xor(w1,1,32);
    float p2 = __shfl_xor(w2,1,32), p3 = __shfl_xor(w3,1,32);
    if ((lane & 1) == 0) {
      *reinterpret_cast<unsigned*>(&Attb[dst])      = pack2bf(w0, p0);
      *reinterpret_cast<unsigned*>(&Attb[dst + 16]) = pack2bf(w1, p1);
      *reinterpret_cast<unsigned*>(&Attb[dst + 32]) = pack2bf(w2, p2);
      *reinterpret_cast<unsigned*>(&Attb[dst + 48]) = pack2bf(w3, p3);
    }
  }
}

// ---------------------------------------------------------------------------
__global__ void unpatchify(const float* __restrict__ Fin, float* __restrict__ Cin)
{
  int idx = blockIdx.x * 256 + threadIdx.x;    // ((b*64+ch)*64+y)*64+x
  if (idx >= BSZ * 64 * 64 * 64) return;
  int x = idx & 63, y = (idx >> 6) & 63, ch = (idx >> 12) & 63, b = idx >> 18;
  int l = (y >> 1) * 32 + (x >> 1);
  Cin[idx] = Fin[((long)(b * SEQ + l)) * 256 + ch * 4 + (y & 1) * 2 + (x & 1)];
}

__global__ void conv3x3_relu16(const float* __restrict__ In, const float* __restrict__ W,
                               const float* __restrict__ Bi, float* __restrict__ Out)
{
  int idx = blockIdx.x * 256 + threadIdx.x;    // 2*16*64*64
  if (idx >= BSZ * 16 * 64 * 64) return;
  int x = idx & 63, y = (idx >> 6) & 63, oc = (idx >> 12) & 15, b = idx >> 16;
  float acc = Bi[oc];
  for (int ic = 0; ic < 64; ++ic) {
    const float* ip = In + ((long)(b * 64 + ic)) * 4096;
    const float* wp = W + (oc * 64 + ic) * 9;
#pragma unroll
    for (int ky = 0; ky < 3; ++ky) {
      int yy = y + ky - 1;
      if ((unsigned)yy >= 64u) continue;
#pragma unroll
      for (int kx = 0; kx < 3; ++kx) {
        int xx = x + kx - 1;
        if ((unsigned)xx >= 64u) continue;
        acc += ip[yy * 64 + xx] * wp[ky * 3 + kx];
      }
    }
  }
  Out[idx] = fmaxf(acc, 0.f);
}

__global__ void conv3x3_out3(const float* __restrict__ In, const float* __restrict__ W,
                             const float* __restrict__ Bi, float* __restrict__ Out)
{
  int idx = blockIdx.x * 256 + threadIdx.x;    // 2*3*64*64
  if (idx >= BSZ * 3 * 64 * 64) return;
  int x = idx & 63, y = (idx >> 6) & 63;
  int oc = (idx >> 12) % 3, b = (idx >> 12) / 3;
  float acc = Bi[oc];
  for (int ic = 0; ic < 16; ++ic) {
    const float* ip = In + ((long)(b * 16 + ic)) * 4096;
    const float* wp = W + (oc * 16 + ic) * 9;
#pragma unroll
    for (int ky = 0; ky < 3; ++ky) {
      int yy = y + ky - 1;
      if ((unsigned)yy >= 64u) continue;
#pragma unroll
      for (int kx = 0; kx < 3; ++kx) {
        int xx = x + kx - 1;
        if ((unsigned)xx >= 64u) continue;
        acc += ip[yy * 64 + xx] * wp[ky * 3 + kx];
      }
    }
  }
  Out[idx] = acc;
}

// ---------------------------------------------------------------------------
extern "C" void kernel_launch(void* const* d_in, const int* in_sizes, int n_in,
                              void* d_out, int out_size, void* d_ws, size_t ws_size,
                              hipStream_t stream)
{
  (void)in_sizes; (void)n_in; (void)out_size; (void)ws_size;
  const float* x     = (const float*)d_in[0];
  const float* qkv_w = (const float*)d_in[1];
  const float* qkv_b = (const float*)d_in[2];
  const float* proj_w= (const float*)d_in[3];
  const float* proj_b= (const float*)d_in[4];
  const float* nq_g  = (const float*)d_in[5];
  const float* nq_b  = (const float*)d_in[6];
  const float* nk_g  = (const float*)d_in[7];
  const float* nk_b  = (const float*)d_in[8];
  const float* n1_g  = (const float*)d_in[9];
  const float* n1_b  = (const float*)d_in[10];
  const float* n2_g  = (const float*)d_in[11];
  const float* n2_b  = (const float*)d_in[12];
  const float* fc1_w = (const float*)d_in[13];
  const float* fc1_b = (const float*)d_in[14];
  const float* fc2_w = (const float*)d_in[15];
  const float* fc2_b = (const float*)d_in[16];
  const float* fin_w = (const float*)d_in[17];
  const float* fin_b = (const float*)d_in[18];
  const float* c1_w  = (const float*)d_in[19];
  const float* c1_b  = (const float*)d_in[20];
  const float* c2_w  = (const float*)d_in[21];
  const float* c2_b  = (const float*)d_in[22];

  char* w = (char*)d_ws;
  auto alloc = [&](size_t bytes) { char* p = w; w += (bytes + 255) & ~(size_t)255; return p; };
  float*  xbuf  = (float*) alloc(sizeof(float)  * ROWS * CHN);
  float*  qkv   = (float*) alloc(sizeof(float)  * ROWS * 3 * CHN);
  float*  fino  = (float*) alloc(sizeof(float)  * ROWS * 256);
  float*  cin   = (float*) alloc(sizeof(float)  * BSZ * 64 * 64 * 64);
  float*  c1o   = (float*) alloc(sizeof(float)  * BSZ * 16 * 64 * 64);
  ushort* hnb   = (ushort*)alloc(sizeof(ushort) * ROWS * CHN);
  ushort* qb    = (ushort*)alloc(sizeof(ushort) * ROWS * CHN);
  ushort* kb    = (ushort*)alloc(sizeof(ushort) * ROWS * CHN);
  ushort* vb    = (ushort*)alloc(sizeof(ushort) * ROWS * CHN);
  ushort* attb  = (ushort*)alloc(sizeof(ushort) * ROWS * CHN);
  ushort* mlphb = (ushort*)alloc(sizeof(ushort) * ROWS * 4 * CHN);
  ushort* qkvWt = (ushort*)alloc(sizeof(ushort) * (size_t)NB * 3 * CHN * CHN);
  ushort* projWt= (ushort*)alloc(sizeof(ushort) * (size_t)NB * CHN * CHN);
  ushort* fc1Wt = (ushort*)alloc(sizeof(ushort) * (size_t)NB * 4 * CHN * CHN);
  ushort* fc2Wt = (ushort*)alloc(sizeof(ushort) * (size_t)NB * 4 * CHN * CHN);
  ushort* finWt = (ushort*)alloc(sizeof(ushort) * 256 * CHN);

  // one-time (per-call) weight transpose+convert: ~300 MB HBM traffic ~= 13us
  for (int i = 0; i < NB; ++i) {
    transpose_f32_to_bf16<<<dim3(CHN/32, 3*CHN/32), 256, 0, stream>>>(
        qkv_w + (size_t)i*CHN*3*CHN, qkvWt + (size_t)i*3*CHN*CHN, CHN, 3*CHN);
    transpose_f32_to_bf16<<<dim3(CHN/32, CHN/32), 256, 0, stream>>>(
        proj_w + (size_t)i*CHN*CHN, projWt + (size_t)i*CHN*CHN, CHN, CHN);
    transpose_f32_to_bf16<<<dim3(CHN/32, 4*CHN/32), 256, 0, stream>>>(
        fc1_w + (size_t)i*CHN*4*CHN, fc1Wt + (size_t)i*4*CHN*CHN, CHN, 4*CHN);
    transpose_f32_to_bf16<<<dim3(4*CHN/32, CHN/32), 256, 0, stream>>>(
        fc2_w + (size_t)i*4*CHN*CHN, fc2Wt + (size_t)i*4*CHN*CHN, 4*CHN, CHN);
  }
  transpose_f32_to_bf16<<<dim3(CHN/32, 256/32), 256, 0, stream>>>(fin_w, finWt, CHN, 256);

  add_pos_embed<<<ROWS, 256, 0, stream>>>(x, xbuf);

  for (int i = 0; i < NB; ++i) {
    layernorm_rows<<<ROWS/8, 256, 0, stream>>>(xbuf, n1_g + i*CHN, n1_b + i*CHN, hnb, ROWS, 1e-5f);
    gemm_bf16<<<dim3(ROWS/64, (3*CHN)/128), 256, 0, stream>>>(
        hnb, qkvWt + (size_t)i*3*CHN*CHN, qkv_b + (size_t)i*3*CHN, nullptr, qkv, nullptr,
        ROWS, 3*CHN, CHN, 0);
    qkv_split_qknorm<<<(ROWS*NH)/8, 256, 0, stream>>>(
        qkv, nq_g + i*HD, nq_b + i*HD, nk_g + i*HD, nk_b + i*HD, qb, kb, vb);
    flash_attn<<<dim3(SEQ/128, BSZ*NH), 256, 0, stream>>>(qb, kb, vb, attb);
    gemm_bf16<<<dim3(ROWS/64, CHN/128), 256, 0, stream>>>(
        attb, projWt + (size_t)i*CHN*CHN, proj_b + (size_t)i*CHN, xbuf, xbuf, nullptr,
        ROWS, CHN, CHN, 0);
    layernorm_rows<<<ROWS/8, 256, 0, stream>>>(xbuf, n2_g + i*CHN, n2_b + i*CHN, hnb, ROWS, 1e-5f);
    gemm_bf16<<<dim3(ROWS/64, (4*CHN)/128), 256, 0, stream>>>(
        hnb, fc1Wt + (size_t)i*4*CHN*CHN, fc1_b + (size_t)i*4*CHN, nullptr, nullptr, mlphb,
        ROWS, 4*CHN, CHN, 1);
    gemm_bf16<<<dim3(ROWS/64, CHN/128), 256, 0, stream>>>(
        mlphb, fc2Wt + (size_t)i*4*CHN*CHN, fc2_b + (size_t)i*CHN, xbuf, xbuf, nullptr,
        ROWS, CHN, 4*CHN, 0);
  }

  layernorm_rows<<<ROWS/8, 256, 0, stream>>>(xbuf, nullptr, nullptr, hnb, ROWS, 1e-6f);
  gemm_bf16<<<dim3(ROWS/64, 256/128), 256, 0, stream>>>(
      hnb, finWt, fin_b, nullptr, fino, nullptr, ROWS, 256, CHN, 0);
  unpatchify<<<(BSZ*64*64*64)/256, 256, 0, stream>>>(fino, cin);
  conv3x3_relu16<<<(BSZ*16*64*64)/256, 256, 0, stream>>>(cin, c1_w, c1_b, c1o);
  conv3x3_out3<<<(BSZ*3*64*64 + 255)/256, 256, 0, stream>>>(c1o, c2_w, c2_b, (float*)d_out);
}
